// Block2_90263032692810
// MI455X (gfx1250) — matrix operand order repaired and checked by reading, not compiled, not run on hardware
//
#include <hip/hip_runtime.h>
#include <math.h>

// ---------------- problem constants (from reference) ----------------
#define BATCH   2
#define CH      192
#define HH      64
#define WW      64
#define LSEQ    (HH * WW)          // 4096
#define DMODEL  192
#define DINNER  384
#define DSTATE  16
#define DCONV   4
#define DTR     12
#define NLAYERS 4
#define XPN_PAD 64                 // x_proj N padded 44 -> 64
#define DTK_PAD 32                 // dt_proj K padded 12 -> 32

// ---------------- WMMA types ----------------
typedef __attribute__((ext_vector_type(16))) __bf16 v16bf;
typedef __attribute__((ext_vector_type(8)))  float  v8f;
typedef __attribute__((ext_vector_type(8)))  unsigned int v8u;
typedef __attribute__((ext_vector_type(4)))  int v4i;
typedef unsigned short u16;

// async global->LDS staging (CDNA5 ASYNCcnt path), with safe fallback.
// Probe-verified signature: (v4i addrspace(1)*, v4i addrspace(3)*, imm, imm)
#if __has_builtin(__builtin_amdgcn_global_load_async_to_lds_b128)
#define USE_ASYNC_LDS 1
typedef __attribute__((address_space(1))) v4i gbl_v4i;
typedef __attribute__((address_space(3))) v4i lds_v4i;
#else
#define USE_ASYNC_LDS 0
#endif

__device__ __forceinline__ void wait_async_zero() {
#if USE_ASYNC_LDS
#if __has_builtin(__builtin_amdgcn_s_wait_asynccnt)
  __builtin_amdgcn_s_wait_asynccnt(0);
#else
  asm volatile("s_wait_asynccnt 0x0" ::: "memory");
#endif
#endif
}

__device__ __forceinline__ u16 f32_to_bf16(float f) {
  unsigned int u = __builtin_bit_cast(unsigned int, f);
  unsigned int lsb = (u >> 16) & 1u;
  u += 0x7fffu + lsb;                       // round-to-nearest-even
  return (u16)(u >> 16);
}
__device__ __forceinline__ float silu_f(float v) {
  return v / (1.0f + __expf(-v));
}

// =====================================================================
// bf16 WMMA GEMM:  C[M,N](f32) = A[M,K](bf16,lda) * W[N,K](bf16)^T
// tile 128x64, BK=32, 256 thr = 8 wave32 waves, 16x64 strip per wave,
// double-buffered LDS, async global->LDS staging (ASYNCcnt) when the
// toolchain exposes it; 1 barrier per K-step either way.
// =====================================================================
#define BM 128
#define BN 64
#define BK 32
#define LDS_P 40   // bf16 elems per LDS row (32 + 8 pad)

__global__ __launch_bounds__(256) void gemm_bf16_wmma(
    const u16* __restrict__ A, int lda,      // [M,K] row-major bf16
    const u16* __restrict__ Wt,              // [N,K] row-major bf16 (ldw == K)
    float* __restrict__ C,
    int M, int N, int K)
{
  __shared__ u16 As[2][BM * LDS_P];
  __shared__ u16 Bs[2][BN * LDS_P];

  const int tid  = threadIdx.x;
  const int lane = tid & 31;
  const int wave = tid >> 5;
  const int m0 = blockIdx.x * BM;
  const int n0 = blockIdx.y * BN;

  // staging coordinates (branch-free; every tile is fully in-bounds)
  const int arow  = tid >> 1;               // 128 rows, 2 thr/row
  const int akoff = (tid & 1) * 16;         // 16 bf16 = 32B each
  const int brow  = tid >> 2;               // 64 rows, 4 thr/row
  const int bkoff = (tid & 3) * 8;          // 8 bf16 = 16B each
  const u16* agp = A  + (size_t)(m0 + arow) * lda + akoff;
  const u16* bgp = Wt + (size_t)(n0 + brow) * K   + bkoff;

  auto stage = [&](int buf, int k0) {
    __builtin_prefetch((const char*)(agp + k0) + BK * 2, 0, 1);  // next K slab
#if USE_ASYNC_LDS
    __builtin_amdgcn_global_load_async_to_lds_b128(
        (gbl_v4i*)(agp + k0),
        (lds_v4i*)&As[buf][arow * LDS_P + akoff], 0, 0);
    __builtin_amdgcn_global_load_async_to_lds_b128(
        (gbl_v4i*)(agp + k0 + 8),
        (lds_v4i*)&As[buf][arow * LDS_P + akoff + 8], 0, 0);
    __builtin_amdgcn_global_load_async_to_lds_b128(
        (gbl_v4i*)(bgp + k0),
        (lds_v4i*)&Bs[buf][brow * LDS_P + bkoff], 0, 0);
#else
    const uint4* s0 = (const uint4*)(agp + k0);
    uint4 a0 = s0[0], a1 = s0[1];
    uint4* d0 = (uint4*)&As[buf][arow * LDS_P + akoff];
    d0[0] = a0; d0[1] = a1;
    const uint4* s1 = (const uint4*)(bgp + k0);
    *(uint4*)&Bs[buf][brow * LDS_P + bkoff] = *s1;
#endif
  };

  v8f acc[4];
#pragma unroll
  for (int j = 0; j < 4; ++j)
#pragma unroll
    for (int r = 0; r < 8; ++r) acc[j][r] = 0.0f;

  const int mrow = lane & 15;
  const int kbA  = (lane < 16) ? 0 : 8;     // A frag K pairs (ISA layout)
  const int kbB  = (lane < 16) ? 0 : 8;     // B frag K pairs (same as A)

  const int nk = K / BK;
  stage(0, 0);

  for (int kt = 0; kt < nk; ++kt) {
    wait_async_zero();                      // tile (kt) staging complete
    __syncthreads();
    const int cur = kt & 1;
    if (kt + 1 < nk) stage(cur ^ 1, (kt + 1) * BK);

    // ---- A fragment (16x32, 16-bit A layout: K 0..7/16..23 vs 8..15/24..31)
    v8u au;
    {
      const u16* ab = &As[cur][(wave * 16 + mrow) * LDS_P];
#pragma unroll
      for (int v = 0; v < 8; ++v) {
        const int k = (v < 4) ? (kbA + 2 * v) : (kbA + 16 + 2 * (v - 4));
        au[v] = *(const unsigned int*)(ab + k);
      }
    }
    const v16bf afrag = __builtin_bit_cast(v16bf, au);

#pragma unroll
    for (int j = 0; j < 4; ++j) {
      v8u bu;
      const u16* bb = &Bs[cur][(j * 16 + mrow) * LDS_P];
#pragma unroll
      for (int v = 0; v < 8; ++v) bu[v] = *(const unsigned int*)(bb + ((v < 4) ? (kbB + 2 * v) : (kbB + 16 + 2 * (v - 4))));
      const v16bf bfrag = __builtin_bit_cast(v16bf, bu);
      acc[j] = __builtin_amdgcn_wmma_f32_16x16x32_bf16(
                   false, afrag, false, bfrag, (short)0, acc[j], false, false);
    }
  }

  // ---- store (C/D layout: VGPR r, lane<16 -> M=r, lane>=16 -> M=r+8)
  const int rbase = m0 + wave * 16 + ((lane < 16) ? 0 : 8);
#pragma unroll
  for (int j = 0; j < 4; ++j) {
    const int col = n0 + j * 16 + mrow;
#pragma unroll
    for (int r = 0; r < 8; ++r)
      C[(size_t)(rbase + r) * N + col] = acc[j][r];
  }
}

// =====================================================================
// bf16 conversion / padding kernels
// =====================================================================
__global__ void cvt_bf16_kernel(const float* __restrict__ src,
                                u16* __restrict__ dst, long n4) {
  long i = (long)blockIdx.x * blockDim.x + threadIdx.x;
  if (i >= n4) return;
  const float4 v = ((const float4*)src)[i];
  uint2 o;
  o.x = (unsigned)f32_to_bf16(v.x) | ((unsigned)f32_to_bf16(v.y) << 16);
  o.y = (unsigned)f32_to_bf16(v.z) | ((unsigned)f32_to_bf16(v.w) << 16);
  ((uint2*)dst)[i] = o;
}

// x_proj weights: [L][44][384] f32 -> [L][64][384] bf16, zero-padded rows
__global__ void xproj_pad_kernel(const float* __restrict__ src, u16* __restrict__ dst) {
  long i = (long)blockIdx.x * blockDim.x + threadIdx.x;
  const long n = (long)NLAYERS * XPN_PAD * DINNER;
  if (i >= n) return;
  const int k = i % DINNER;
  const int r = (i / DINNER) % XPN_PAD;
  const int l = i / (DINNER * XPN_PAD);
  const float v = (r < 44) ? src[((long)l * 44 + r) * DINNER + k] : 0.0f;
  dst[i] = f32_to_bf16(v);
}

// dt_proj weights: [L][384][12] f32 -> [L][384][32] bf16, zero-padded cols
__global__ void dtw_pad_kernel(const float* __restrict__ src, u16* __restrict__ dst) {
  long i = (long)blockIdx.x * blockDim.x + threadIdx.x;
  const long n = (long)NLAYERS * DINNER * DTK_PAD;
  if (i >= n) return;
  const int k = i % DTK_PAD;
  const int r = (i / DTK_PAD) % DINNER;
  const int l = i / (DTK_PAD * DINNER);
  const float v = (k < DTR) ? src[((long)l * DINNER + r) * DTR + k] : 0.0f;
  dst[i] = f32_to_bf16(v);
}

// x_dbl activations: dt columns 0..11 of [8192][64] f32 -> [8192][32] bf16 padded
__global__ void xdbl_pad_kernel(const float* __restrict__ src, u16* __restrict__ dst) {
  long i = (long)blockIdx.x * blockDim.x + threadIdx.x;
  const long n = (long)BATCH * LSEQ * DTK_PAD;
  if (i >= n) return;
  const int k = i % DTK_PAD;
  const long r = i / DTK_PAD;
  const float v = (k < DTR) ? src[r * XPN_PAD + k] : 0.0f;
  dst[i] = f32_to_bf16(v);
}

// =====================================================================
// LN kernels
// =====================================================================
// LN1 over W fused with [B,C,H,W] -> [B,L,C] transpose (f32 out)
__global__ __launch_bounds__(64) void ln1_transpose_kernel(
    const float* __restrict__ x, const float* __restrict__ w,
    const float* __restrict__ b, float* __restrict__ m)
{
  __shared__ float red[64];
  const int row = blockIdx.x;               // b*C*H + c*H + h
  const int t   = threadIdx.x;
  const int h   = row % HH;
  const int c   = (row / HH) % CH;
  const int bb  = row / (HH * CH);
  const float v = x[(long)row * WW + t];
  red[t] = v; __syncthreads();
  for (int s = 32; s > 0; s >>= 1) { if (t < s) red[t] += red[t + s]; __syncthreads(); }
  const float mu = red[0] * (1.0f / WW); __syncthreads();
  const float dv = v - mu;
  red[t] = dv * dv; __syncthreads();
  for (int s = 32; s > 0; s >>= 1) { if (t < s) red[t] += red[t + s]; __syncthreads(); }
  const float rstd = rsqrtf(red[0] * (1.0f / WW) + 1e-5f);
  m[((long)bb * LSEQ + h * WW + t) * CH + c] = dv * rstd * w[t] + b[t];
}

// generic row LN (D <= 256), writes bf16 (feeds a WMMA GEMM)
__global__ __launch_bounds__(256) void rowln_bf16_kernel(
    const float* __restrict__ in, u16* __restrict__ out16,
    const float* __restrict__ w, const float* __restrict__ b,
    int rows, int D)
{
  __shared__ float red[256];
  const int row = blockIdx.x;
  const int t = threadIdx.x;
  const float v = (t < D) ? in[(long)row * D + t] : 0.0f;
  red[t] = v; __syncthreads();
  for (int s = 128; s > 0; s >>= 1) { if (t < s) red[t] += red[t + s]; __syncthreads(); }
  const float mu = red[0] / (float)D; __syncthreads();
  const float dv = (t < D) ? (v - mu) : 0.0f;
  red[t] = dv * dv; __syncthreads();
  for (int s = 128; s > 0; s >>= 1) { if (t < s) red[t] += red[t + s]; __syncthreads(); }
  const float rstd = rsqrtf(red[0] / (float)D + 1e-5f);
  if (t < D) out16[(long)row * D + t] = f32_to_bf16(dv * rstd * w[t] + b[t]);
}

// =====================================================================
// elementwise / sequence kernels
// =====================================================================
__global__ void resid_kernel(const float* a, const float* b, float* out,
                             long n, int has_b) {
  long i = (long)blockIdx.x * blockDim.x + threadIdx.x;
  if (i >= n) return;
  out[i] = a[i] + (has_b ? b[i] : 0.0f);
}

__global__ void flip_kernel(const float* src, float* dst) {
  long i = (long)blockIdx.x * blockDim.x + threadIdx.x;
  const long n = (long)BATCH * LSEQ * CH;
  if (i >= n) return;
  const int c = i % CH;
  const long bl = i / CH;
  const int l = bl % LSEQ, b = bl / LSEQ;
  dst[i] = src[((long)b * LSEQ + (LSEQ - 1 - l)) * CH + c];
}

__global__ void combine_odd_kernel(float* m, float* res,
                                   const float* mf, const float* rf) {
  long i = (long)blockIdx.x * blockDim.x + threadIdx.x;
  const long n = (long)BATCH * LSEQ * CH;
  if (i >= n) return;
  const int c = i % CH;
  const long bl = i / CH;
  const int l = bl % LSEQ, b = bl / LSEQ;
  const long fi = ((long)b * LSEQ + (LSEQ - 1 - l)) * CH + c;
  m[i]   += mf[fi];
  res[i] += rf[fi];
}

// causal depthwise conv1d (kernel 4, left pad 3) + bias + SiLU
// writes f32 (for scan) and bf16 (for x_proj GEMM)
__global__ void conv_silu_kernel(const float* __restrict__ xz,
                                 const float* __restrict__ cw,
                                 const float* __restrict__ cb,
                                 float* __restrict__ xc,
                                 u16* __restrict__ xc16) {
  long i = (long)blockIdx.x * blockDim.x + threadIdx.x;
  const long n = (long)BATCH * LSEQ * DINNER;
  if (i >= n) return;
  const int d = i % DINNER;
  const long bl = i / DINNER;
  const int l = bl % LSEQ, b = bl / LSEQ;
  const float* base = xz + (long)b * LSEQ * (2 * DINNER) + d;
  float acc = cb[d];
#pragma unroll
  for (int j = 0; j < DCONV; ++j) {
    const int ls = l - (DCONV - 1) + j;
    if (ls >= 0) acc += cw[d * DCONV + j] * base[(long)ls * (2 * DINNER)];
  }
  const float s = silu_f(acc);
  xc[i] = s;
  xc16[i] = f32_to_bf16(s);
}

__global__ void softplus_bias_kernel(float* __restrict__ delta,
                                     const float* __restrict__ dtb) {
  long i = (long)blockIdx.x * blockDim.x + threadIdx.x;
  const long n = (long)BATCH * LSEQ * DINNER;
  if (i >= n) return;
  const float v = delta[i] + dtb[i % DINNER];
  delta[i] = (v > 20.0f) ? v : log1pf(__expf(v));
}

// selective scan: one LANE per (b, d, s) state element; y-reduction over the
// 16 state lanes via shfl_xor butterfly (lane-permute ops).
__global__ __launch_bounds__(256) void scan_kernel(
    const float* __restrict__ xc, const float* __restrict__ delta,
    const float* __restrict__ xdbl, const float* __restrict__ A_log,
    const float* __restrict__ ssm_D, float* __restrict__ y)
{
  const int tid = blockIdx.x * blockDim.x + threadIdx.x;
  if (tid >= BATCH * DINNER * DSTATE) return;
  const int s  = tid & (DSTATE - 1);
  const int ch = tid >> 4;                  // (b, d)
  const int b  = ch / DINNER, d = ch % DINNER;
  const float a  = -__expf(A_log[d * DSTATE + s]);
  const float Dv = ssm_D[d];
  float h = 0.0f;
  for (int l = 0; l < LSEQ; ++l) {
    const long ro = (long)b * LSEQ + l;
    const float u  = xc[ro * DINNER + d];
    const float dt = delta[ro * DINNER + d];
    const float Bm = xdbl[ro * XPN_PAD + DTR + s];
    const float Cm = xdbl[ro * XPN_PAD + DTR + DSTATE + s];
    h = h * __expf(dt * a) + dt * u * Bm;
    float acc = h * Cm;
#pragma unroll
    for (int m = 8; m >= 1; m >>= 1) acc += __shfl_xor(acc, m, DSTATE);
    if (s == 0) y[ro * DINNER + d] = acc + u * Dv;
  }
}

// g16 = bf16( y * silu(z) )   (feeds out_proj GEMM)
__global__ void gate_kernel(const float* __restrict__ y,
                            const float* __restrict__ xz,
                            u16* __restrict__ g16) {
  long i = (long)blockIdx.x * blockDim.x + threadIdx.x;
  const long n = (long)BATCH * LSEQ * DINNER;
  if (i >= n) return;
  const int d = i % DINNER;
  const long bl = i / DINNER;
  const float z = xz[bl * (2 * DINNER) + DINNER + d];
  g16[i] = f32_to_bf16(y[i] * silu_f(z));
}

// M4[b,c,h,w] = m[b, h*W+w, c] + x[b,c,h,w]
__global__ void transpose_add_kernel(const float* __restrict__ m,
                                     const float* __restrict__ x,
                                     float* __restrict__ m4) {
  long i = (long)blockIdx.x * blockDim.x + threadIdx.x;
  const long n = (long)BATCH * CH * HH * WW;
  if (i >= n) return;
  const int hw = i % (HH * WW);
  const int c  = (i / (HH * WW)) % CH;
  const int b  = i / ((long)CH * HH * WW);
  m4[i] = m[((long)b * LSEQ + hw) * CH + c] + x[i];
}

// h116 = bf16( gelu_exact(h1 + b1) )   (feeds mlp2 GEMM)
__global__ void bias_gelu_kernel(const float* __restrict__ h1,
                                 const float* __restrict__ b1,
                                 u16* __restrict__ out16, long n, int N) {
  long i = (long)blockIdx.x * blockDim.x + threadIdx.x;
  if (i >= n) return;
  const float v = h1[i] + b1[i % N];
  out16[i] = f32_to_bf16(0.5f * v * (1.0f + erff(v * 0.70710678118654752440f)));
}

__global__ void final_add_kernel(const float* __restrict__ h2,
                                 const float* __restrict__ b2,
                                 const float* __restrict__ m4,
                                 float* __restrict__ out) {
  long i = (long)blockIdx.x * blockDim.x + threadIdx.x;
  const long n = (long)BATCH * CH * HH * WW;
  if (i >= n) return;
  out[i] = h2[i] + b2[i % WW] + m4[i];
}

// =====================================================================
// host-side orchestration
// =====================================================================
static inline void launch_gemm(const u16* A, int lda, const u16* Wt,
                               float* C, int M, int N, int K, hipStream_t s) {
  dim3 g((M + BM - 1) / BM, (N + BN - 1) / BN);
  gemm_bf16_wmma<<<g, 256, 0, s>>>(A, lda, Wt, C, M, N, K);
}
static inline int eb(long n) { return (int)((n + 255) / 256); }

extern "C" void kernel_launch(void* const* d_in, const int* in_sizes, int n_in,
                              void* d_out, int out_size, void* d_ws, size_t ws_size,
                              hipStream_t stream) {
  (void)in_sizes; (void)n_in; (void)out_size; (void)ws_size;
  const float* x        = (const float*)d_in[0];
  const float* ln1_w    = (const float*)d_in[1];
  const float* ln1_b    = (const float*)d_in[2];
  const float* norm_w   = (const float*)d_in[3];
  const float* norm_b   = (const float*)d_in[4];
  const float* in_proj  = (const float*)d_in[5];
  const float* conv_w   = (const float*)d_in[6];
  const float* conv_b   = (const float*)d_in[7];
  const float* x_proj   = (const float*)d_in[8];
  const float* dt_projw = (const float*)d_in[9];
  const float* dt_projb = (const float*)d_in[10];
  const float* A_log    = (const float*)d_in[11];
  const float* ssm_D    = (const float*)d_in[12];
  const float* out_proj = (const float*)d_in[13];
  const float* ln2_w    = (const float*)d_in[14];
  const float* ln2_b    = (const float*)d_in[15];
  const float* mlp_w1   = (const float*)d_in[16];
  const float* mlp_b1   = (const float*)d_in[17];
  const float* mlp_w2   = (const float*)d_in[18];
  const float* mlp_b2   = (const float*)d_in[19];
  float* out = (float*)d_out;

  // -------- workspace layout --------
  const long SZ_BLC = (long)BATCH * LSEQ * CH;          // 1,572,864
  const long SZ_NDI = (long)BATCH * LSEQ * DINNER;      // 3,145,728
  float* w = (float*)d_ws;
  float* M_    = w; w += SZ_BLC;                        // running m [B,L,C]
  float* RES   = w; w += SZ_BLC;
  float* HB    = w; w += SZ_BLC;                        // flip buf / M4 reuse
  float* RIN   = w; w += SZ_BLC;
  float* ROUT  = w; w += SZ_BLC;
  float* MIX   = w; w += SZ_BLC;                        // mixer out / h2 reuse
  float* XZ    = w; w += 2 * SZ_NDI;                    // xz / h1 reuse
  float* XC    = w; w += SZ_NDI;
  float* DELTA = w; w += SZ_NDI;
  float* Y     = w; w += SZ_NDI;
  float* XDBL  = w; w += (long)BATCH * LSEQ * XPN_PAD;  // [8192,64] padded
  // bf16 region
  u16* u = (u16*)w;
  u16* W16_IP  = u; u += (long)NLAYERS * 2 * DINNER * DMODEL;  // 589,824
  u16* W16_XP  = u; u += (long)NLAYERS * XPN_PAD * DINNER;     //  98,304
  u16* W16_DT  = u; u += (long)NLAYERS * DINNER * DTK_PAD;     //  49,152
  u16* W16_OP  = u; u += (long)NLAYERS * DMODEL * DINNER;      // 294,912
  u16* W16_M1  = u; u += 4 * WW * WW;                          //  16,384
  u16* W16_M2  = u; u += 4 * WW * WW;                          //  16,384
  u16* HN16    = u; u += SZ_BLC;                    // LN out / LN2(T) reuse
  u16* XC16    = u; u += SZ_NDI;
  u16* XDBL16  = u; u += (long)BATCH * LSEQ * DTK_PAD;
  u16* G16     = u; u += SZ_NDI;
  u16* H116    = u; u += 2 * SZ_NDI;                // 24576*256

  const int  ROWS  = BATCH * LSEQ;                  // 8192 tokens
  const int  MROWS = BATCH * CH * HH;               // 24576 W-rows
  const long NCHW  = (long)BATCH * CH * HH * WW;

  // -------- one-time weight conversion to bf16 (padded where needed) ----
  cvt_bf16_kernel<<<eb((long)NLAYERS * 2 * DINNER * DMODEL / 4), 256, 0, stream>>>(
      in_proj, W16_IP, (long)NLAYERS * 2 * DINNER * DMODEL / 4);
  xproj_pad_kernel<<<eb((long)NLAYERS * XPN_PAD * DINNER), 256, 0, stream>>>(x_proj, W16_XP);
  dtw_pad_kernel<<<eb((long)NLAYERS * DINNER * DTK_PAD), 256, 0, stream>>>(dt_projw, W16_DT);
  cvt_bf16_kernel<<<eb((long)NLAYERS * DMODEL * DINNER / 4), 256, 0, stream>>>(
      out_proj, W16_OP, (long)NLAYERS * DMODEL * DINNER / 4);
  cvt_bf16_kernel<<<eb(4 * WW * WW / 4), 256, 0, stream>>>(mlp_w1, W16_M1, 4 * WW * WW / 4);
  cvt_bf16_kernel<<<eb(4 * WW * WW / 4), 256, 0, stream>>>(mlp_w2, W16_M2, 4 * WW * WW / 4);

  // -------- LN1 over W + transpose to [B,L,C] --------
  ln1_transpose_kernel<<<BATCH * CH * HH, 64, 0, stream>>>(x, ln1_w, ln1_b, M_);

  for (int layer = 0; layer < NLAYERS; ++layer) {
    const bool odd = (layer & 1) != 0;
    const u16* ipw = W16_IP + (long)layer * 2 * DINNER * DMODEL;
    const u16* xpw = W16_XP + (long)layer * XPN_PAD * DINNER;
    const u16* dpw = W16_DT + (long)layer * DINNER * DTK_PAD;
    const u16* opw = W16_OP + (long)layer * DMODEL * DINNER;
    const float* cw  = conv_w   + (long)layer * DINNER * DCONV;
    const float* cb  = conv_b   + (long)layer * DINNER;
    const float* dpb = dt_projb + (long)layer * DINNER;
    const float* alg = A_log    + (long)layer * DINNER * DSTATE;
    const float* sdv = ssm_D    + (long)layer * DINNER;
    const float* nw  = norm_w   + (long)layer * DMODEL;
    const float* nb  = norm_b   + (long)layer * DMODEL;

    const float* resout;
    if (odd) {
      flip_kernel<<<eb(SZ_BLC), 256, 0, stream>>>(M_, HB);
      flip_kernel<<<eb(SZ_BLC), 256, 0, stream>>>(RES, RIN);
      resid_kernel<<<eb(SZ_BLC), 256, 0, stream>>>(HB, RIN, ROUT, SZ_BLC, 1);
      resout = ROUT;
    } else {
      resid_kernel<<<eb(SZ_BLC), 256, 0, stream>>>(M_, RES, RES, SZ_BLC,
                                                   layer == 0 ? 0 : 1);
      resout = RES;
    }
    rowln_bf16_kernel<<<ROWS, 256, 0, stream>>>(resout, HN16, nw, nb, ROWS, DMODEL);

    launch_gemm(HN16, DMODEL, ipw, XZ, ROWS, 2 * DINNER, DMODEL, stream);  // in_proj
    conv_silu_kernel<<<eb(SZ_NDI), 256, 0, stream>>>(XZ, cw, cb, XC, XC16);
    launch_gemm(XC16, DINNER, xpw, XDBL, ROWS, XPN_PAD, DINNER, stream);   // x_proj (N padded)
    xdbl_pad_kernel<<<eb((long)ROWS * DTK_PAD), 256, 0, stream>>>(XDBL, XDBL16);
    launch_gemm(XDBL16, DTK_PAD, dpw, DELTA, ROWS, DINNER, DTK_PAD, stream); // dt_proj (K padded)
    softplus_bias_kernel<<<eb(SZ_NDI), 256, 0, stream>>>(DELTA, dpb);
    scan_kernel<<<(BATCH * DINNER * DSTATE + 255) / 256, 256, 0, stream>>>(
        XC, DELTA, XDBL, alg, sdv, Y);
    gate_kernel<<<eb(SZ_NDI), 256, 0, stream>>>(Y, XZ, G16);
    launch_gemm(G16, DINNER, opw, odd ? MIX : M_, ROWS, DMODEL, DINNER, stream);
    if (odd)
      combine_odd_kernel<<<eb(SZ_BLC), 256, 0, stream>>>(M_, RES, MIX, ROUT);
  }

  // -------- tail: residual + LN2 + MLP over W --------
  float* M4 = HB;
  float* H1 = XZ;
  float* H2 = MIX;
  u16*   T16 = HN16;

  transpose_add_kernel<<<eb(NCHW), 256, 0, stream>>>(M_, x, M4);
  rowln_bf16_kernel<<<MROWS, 256, 0, stream>>>(M4, T16, ln2_w, ln2_b, MROWS, WW);
  launch_gemm(T16, WW, W16_M1, H1, MROWS, 4 * WW, WW, stream);             // mlp1
  bias_gelu_kernel<<<eb((long)MROWS * 4 * WW), 256, 0, stream>>>(
      H1, mlp_b1, H116, (long)MROWS * 4 * WW, 4 * WW);
  launch_gemm(H116, 4 * WW, W16_M2, H2, MROWS, WW, 4 * WW, stream);        // mlp2
  final_add_kernel<<<eb(NCHW), 256, 0, stream>>>(H2, mlp_b2, M4, out);
}
